// InpaintingBlock_70609262346829
// MI455X (gfx1250) — compile-verified, matrix operand order checked by reading, not run on hardware
//
#include <hip/hip_runtime.h>

// ---------------------------------------------------------------------------
// Inpainting diffusion: 20 temporally-blocked stencil steps in one kernel.
// Per 16x16 subtile per step: Y = A_v * B_v + A_h * B_h via two
// v_wmma_f32_16x16x32_f16 (banded constant operand carries the stencil,
// K=18 of 32 used so the +-1 halo terms live inside the matrix product).
// State in LDS: f16 ping-pong buffers + exact f32 originals for pinning.
// Interior workgroups fill the f32 region via the Tensor Data Mover
// (tensor_load_to_lds + s_wait_tensorcnt); image-edge workgroups use a
// scalar reflect-indexed fill.
// ---------------------------------------------------------------------------

typedef __attribute__((ext_vector_type(16))) _Float16     v16h;
typedef __attribute__((ext_vector_type(8)))  float        v8f;
typedef __attribute__((ext_vector_type(4)))  unsigned int u32x4;
typedef __attribute__((ext_vector_type(8)))  int          i32x8;
typedef __attribute__((ext_vector_type(4)))  int          i32x4;

#if __has_builtin(__builtin_amdgcn_tensor_load_to_lds) && \
    __has_builtin(__builtin_amdgcn_s_wait_tensorcnt)
#define HAVE_TDM 1
#else
#define HAVE_TDM 0
#endif

#define IMG_H   2048
#define IMG_W   2048
#define N_IMG   16
#define HALO    20
#define TILE    120                  // output tile per workgroup
#define R       160                  // TILE + 2*HALO, 10x10 WMMA subtiles
#define NT      10                   // R/16
#define NTILES  (NT*NT)              // 100 subtiles per region
#define PITCH_H 165                  // f16 LDS pitch (halves)
#define PITCH_F 160                  // f32 LDS pitch (floats); TDM writes rows packed
#define TPB     18                   // ceil(2048/120) tiles per image dim
#define SMEM_BYTES (2*(R*PITCH_H)*2 + (R*PITCH_F)*4)   // 105600 + 102400 = 208000 B

__device__ __forceinline__ int reflect_idx(int g, int n) {
  g = (g < 0) ? -g : g;                // jnp 'reflect': -1 -> 1
  g = (g >= n) ? (2*n - 2 - g) : g;    // n -> n-2
  return g;
}

__global__ __launch_bounds__(256)
void inpaint_wmma_kernel(const float* __restrict__ x,
                         const float* __restrict__ wstencil,
                         const int*   __restrict__ tsteps,
                         float*       __restrict__ out) {
  extern __shared__ unsigned char smem[];
  _Float16* bufA = (_Float16*)smem;                          // R*PITCH_H halves
  _Float16* bufB = bufA + R*PITCH_H;
  float*    orig = (float*)(smem + (size_t)2*(R*PITCH_H)*2); // R*PITCH_F floats

  const int tid  = threadIdx.x;
  const int lane = tid & 31;
  const int wv   = tid >> 5;     // wave id 0..7
  const int q    = lane & 15;    // row (A) / col (B,C,D) index
  const int hh   = lane >> 4;    // lane half

  const int bid = blockIdx.x;
  const int img = bid / (TPB*TPB);
  const int trc = bid % (TPB*TPB);
  const int ty  = trc / TPB;
  const int tx  = trc % TPB;
  const int gy0 = ty*TILE - HALO;
  const int gx0 = tx*TILE - HALO;
  const float* xi = x + (size_t)img * ((size_t)IMG_H * IMG_W);

  // stencil weights: w[0,0,0,1]=wy (up/down), w[0,0,1,0]=wx (left/right), center w[4]
  const float wy = wstencil[1];
  const float wx = wstencil[3];
  const float wc = wstencil[4];
  const _Float16 hy = (_Float16)wy, hx = (_Float16)wx, hc = (_Float16)wc;
  const _Float16 z  = (_Float16)0.0f;

  // Constant banded fragments. K axis maps to source row/col = subtile_origin-1+k,
  // valid k in [0,18), rest zero-padded.
  // A (16x32) and B (32x16) f16 fragments use the same two-stacked-16x16-tile
  // K mapping per lane: e<8 -> K=8*hh+e ; e>=8 -> K=16+8*hh+(e-8)
  v16h tfa = {};   // vertical pass A: wy @ k==q, wc @ k==q+1, wy @ k==q+2
  v16h tfb = {};   // horizontal pass B: wx @ k==q, wx @ k==q+2
#pragma unroll
  for (int e = 0; e < 16; ++e) {
    const int k = (e < 8) ? (8*hh + e) : (16 + 8*hh + (e - 8));
    tfa[e] = (k == q) ? hy : ((k == q + 1) ? hc : ((k == q + 2) ? hy : z));
    tfb[e] = (k == q) ? hx : ((k == q + 2) ? hx : z);
  }

  // ---- fill region: exact f32 originals + f16 state --------------------------
#if HAVE_TDM
  const bool interior = (gy0 >= 0) && (gx0 >= 0) &&
                        (gy0 + R <= IMG_H) && (gx0 + R <= IMG_W);
  if (interior) {
    // 2D TDM descriptor (ISA 8.3-8.5): 160x160 f32 tile of a 2048-stride tensor.
    if (wv == 0) {
      const unsigned long long ga =
          (unsigned long long)(size_t)(xi + (size_t)gy0 * IMG_W + gx0);
      const unsigned int ldsOff = (unsigned int)(size_t)orig; // low32 = LDS offset
      u32x4 g0;
      g0[0] = 1u;                                        // count=1 (user mode)
      g0[1] = ldsOff;                                    // lds_addr
      g0[2] = (unsigned int)ga;                          // global_addr[31:0]
      g0[3] = ((unsigned int)(ga >> 32) & 0x01FFFFFFu)   // global_addr[56:32]
              | 0x80000000u;                             // type=2 ("image")
      i32x8 g1;
      g1[0] = 0x00020000;                                // data_size=4B, no multicast
      g1[1] = (int)((IMG_W & 0xFFFF) << 16);             // tensor_dim0 lo16
      g1[2] = (int)(((IMG_W >> 16) & 0xFFFF) |
                    ((IMG_H & 0xFFFF) << 16));           // dim0 hi16 | dim1 lo16
      g1[3] = (int)(((IMG_H >> 16) & 0xFFFF) |
                    (((unsigned)R) << 16));              // dim1 hi16 | tile_dim0
      g1[4] = (int)R;                                    // tile_dim1 (tile_dim2=0)
      g1[5] = (int)IMG_W;                                // tensor_dim0_stride lo32
      g1[6] = 0;                                         // stride0 hi | stride1 lo
      g1[7] = 0;
      const i32x4 gz4 = {0, 0, 0, 0};                    // 2D: groups 2/3 unused
      const i32x8 gz8 = {0, 0, 0, 0, 0, 0, 0, 0};
      __builtin_amdgcn_tensor_load_to_lds(g0, g1, gz4, gz4, gz8, 0);
      __builtin_amdgcn_s_wait_tensorcnt(0);
    }
    __syncthreads();                                     // orig[] now valid
    for (int idx = tid; idx < R*R; idx += 256) {         // 25600 = 100*256
      const int rr = idx / R, cc = idx % R;
      bufA[rr*PITCH_H + cc] = (_Float16)orig[rr*PITCH_F + cc];
    }
  } else
#endif
  {
    for (int idx = tid; idx < R*R; idx += 256) {
      const int rr = idx / R, cc = idx % R;
      const int gy = reflect_idx(gy0 + rr, IMG_H);
      const int gx = reflect_idx(gx0 + cc, IMG_W);
      const float v = xi[(size_t)gy*IMG_W + gx];
      orig[rr*PITCH_F + cc] = v;
      bufA[rr*PITCH_H + cc] = (_Float16)v;
    }
  }
  __syncthreads();

  const int T = tsteps[0];
  _Float16* cur = bufA;
  _Float16* nxt = bufB;

  for (int s = 0; s < T; ++s) {
    for (int t = wv; t < NTILES; t += 8) {     // wave-uniform: EXEC all ones at WMMA
      const int sr = (t / NT) * 16;
      const int sc = (t % NT) * 16;

      // B_v: rows sr-1 .. sr+16 of cur at column sc+q
      v16h bv = {};
      {
        const int col = sc + q;
#pragma unroll
        for (int e = 0; e < 8; ++e) {
          int r0 = sr - 1 + 8*hh + e;            // K=8*hh+e -> row sr-1+K
          r0 = (r0 < 0) ? 0 : r0;
          bv[e] = cur[r0*PITCH_H + col];
        }
        const int r16 = (sr + 16 < R) ? (sr + 16) : (R - 1);
        bv[8] = (hh == 0) ? cur[(sr + 15)*PITCH_H + col] : z;  // K=16
        bv[9] = (hh == 0) ? cur[r16*PITCH_H + col]       : z;  // K=17 (K>=18 zero)
      }

      // A_h: row sr+q of cur at columns sc-1 .. sc+16
      v16h ah = {};
      {
        const int rowb = (sr + q) * PITCH_H;
#pragma unroll
        for (int e = 0; e < 8; ++e) {
          int c0 = sc - 1 + 8*hh + e;            // K=8*hh+e -> col sc-1+K
          c0 = (c0 < 0) ? 0 : c0;
          ah[e] = cur[rowb + c0];
        }
        const int c16 = (sc + 16 < R) ? (sc + 16) : (R - 1);
        ah[8] = (hh == 0) ? cur[rowb + sc + 15] : z;   // K=16
        ah[9] = (hh == 0) ? cur[rowb + c16]     : z;   // K=17
      }

      v8f d = {};
      d = __builtin_amdgcn_wmma_f32_16x16x32_f16(false, tfa, false, bv,
                                                 (short)0, d, false, false);
      d = __builtin_amdgcn_wmma_f32_16x16x32_f16(false, ah, false, tfb,
                                                 (short)0, d, false, false);

      // pin known pixels to exact originals, store f16 next state
#pragma unroll
      for (int v = 0; v < 8; ++v) {
        const int M  = v + 8*hh;               // C/D layout: M=v+8*hh, N=q
        const int rr = sr + M, cc = sc + q;
        const float po  = orig[rr*PITCH_F + cc];
        const float val = (po > 0.0f) ? po : d[v];
        nxt[rr*PITCH_H + cc] = (_Float16)val;
      }
    }
    __syncthreads();
    _Float16* tmp = cur; cur = nxt; nxt = tmp;
  }

  // ---- write owned interior (pinned pixels exact from f32 orig) --------------
  float* oi = out + (size_t)img * ((size_t)IMG_H * IMG_W);
  for (int idx = tid; idx < TILE*TILE; idx += 256) {
    const int oy = idx / TILE, ox = idx % TILE;
    const int gy = ty*TILE + oy, gx = tx*TILE + ox;
    if (gy < IMG_H && gx < IMG_W) {
      const int rr = HALO + oy, cc = HALO + ox;
      const float po = orig[rr*PITCH_F + cc];
      oi[(size_t)gy*IMG_W + gx] = (po > 0.0f) ? po : (float)cur[rr*PITCH_H + cc];
    }
  }
}

extern "C" void kernel_launch(void* const* d_in, const int* in_sizes, int n_in,
                              void* d_out, int out_size, void* d_ws, size_t ws_size,
                              hipStream_t stream) {
  (void)in_sizes; (void)n_in; (void)out_size; (void)d_ws; (void)ws_size;
  const float* x  = (const float*)d_in[0];
  const float* w  = (const float*)d_in[1];
  const int*   ts = (const int*)d_in[2];
  float* out = (float*)d_out;

  dim3 grid(N_IMG * TPB * TPB);   // 16 images * 18*18 tiles = 5184 workgroups
  dim3 block(256);                // 8 waves (wave32)
  inpaint_wmma_kernel<<<grid, block, SMEM_BYTES, stream>>>(x, w, ts, out);
}
